// HGCNLayer_4252017623763
// MI455X (gfx1250) — matrix operand order repaired and checked by reading, not compiled, not run on hardware
//
#include <hip/hip_runtime.h>
#include <hip/hip_bf16.h>

typedef __attribute__((ext_vector_type(16))) _Float16 v16h;
typedef __attribute__((ext_vector_type(8)))  float    v8f;

#define HD 64
#define HK 4

union AV { v16h v; _Float16 h[16]; };
union CV { v8f  v; float    f[8];  };

// ---------------------------------------------------------------- zero init
__global__ void __launch_bounds__(256) k_zero(float* out, long long n_out,
                                              float* deg, long long n_deg) {
  long long stride = (long long)gridDim.x * blockDim.x;
  long long i0 = (long long)blockIdx.x * blockDim.x + threadIdx.x;
  for (long long i = i0; i < n_out; i += stride) out[i] = 0.0f;
  for (long long i = i0; i < n_deg; i += stride) deg[i] = 0.0f;
}

// ------------------------------------------- forward edge gather/scatter-add
// 16 lanes per edge, float4 gather, dword f32 atomics into rst (fits in L2).
__global__ void __launch_bounds__(256) k_fwd(const float* __restrict__ feat_src,
                                             const int* __restrict__ es,
                                             const int* __restrict__ ed,
                                             float* rst, float* deg_dst,
                                             float* deg_src, long long nE) {
  long long idx = (long long)blockIdx.x * blockDim.x + threadIdx.x;
  long long e = idx >> 4;
  if (e >= nE) return;
  int q = (int)(idx & 15);
  int s = es[e], d = ed[e];
  const float4 v = *(const float4*)(feat_src + (long long)s * HD + q * 4);
  float* p = rst + (long long)d * HD + q * 4;
  atomicAdd(p + 0, v.x);
  atomicAdd(p + 1, v.y);
  atomicAdd(p + 2, v.z);
  atomicAdd(p + 3, v.w);
  if (q == 0) {
    atomicAdd(deg_dst + d, 1.0f);
    atomicAdd(deg_src + s, 1.0f);
  }
}

// ---------------------------------------------------- backward edge scatter
__global__ void __launch_bounds__(256) k_bwd(const float* __restrict__ rst,
                                             const int* __restrict__ es,
                                             const int* __restrict__ ed,
                                             float* bsrc, long long nE) {
  long long idx = (long long)blockIdx.x * blockDim.x + threadIdx.x;
  long long e = idx >> 4;
  if (e >= nE) return;
  int q = (int)(idx & 15);
  int s = es[e], d = ed[e];
  const float4 v = *(const float4*)(rst + (long long)d * HD + q * 4);
  float* p = bsrc + (long long)s * HD + q * 4;
  atomicAdd(p + 0, v.x);
  atomicAdd(p + 1, v.y);
  atomicAdd(p + 2, v.z);
  atomicAdd(p + 3, v.w);
}

// -------------------------------------------------- src degree normalization
__global__ void __launch_bounds__(256) k_norm(float* bsrc,
                                              const float* __restrict__ deg_src,
                                              long long n) {
  long long i = (long long)blockIdx.x * blockDim.x + threadIdx.x;
  if (i < n) bsrc[i] *= 1.0f / fmaxf(deg_src[i >> 6], 1.0f);
}

// --------------------------------------------------------- WMMA attention
// One wave per 16 dst nodes. scores and pf_att computed exactly with
// v_wmma_f32_16x16x32_f16 (diagonal-block extraction / block-diagonal A).
// rst is finalized in place: rst = (raw + 0.1*tanh(pf_att)) / max(deg,1).
__global__ void __launch_bounds__(256) k_attn(float* rst,
                                              const float* __restrict__ pf,
                                              const float* __restrict__ deg_dst,
                                              int n_dst) {
  __shared__ float    s_rst[8][16][HD + 1];
  __shared__ float    s_sc [8][16][HK];
  __shared__ _Float16 s_at [8][16][HK];
  __shared__ float    s_inv[8][16];

  const int lane = threadIdx.x & 31;
  const int wave = threadIdx.x >> 5;
  const int g    = (blockIdx.x * 8 + wave) * 16;   // base node of this wave
  const int hi   = (lane >= 16) ? 1 : 0;
  const int m16  = lane & 15;

  // stage raw rst tile (16 nodes x 64) into LDS, coalesced; clamp tail reads
  for (int t = 0; t < 32; ++t) {
    int lin = t * 32 + lane;
    int mm = lin >> 6, dd = lin & 63;
    int node = g + mm;
    if (node >= n_dst) node = n_dst - 1;
    s_rst[wave][mm][dd] = rst[(long long)node * HD + dd];
  }
  if (lane < 16) {
    int node = g + lane;
    float dg = (node < n_dst) ? deg_dst[node] : 1.0f;
    s_inv[wave][lane] = 1.0f / fmaxf(dg, 1.0f);
  }
  __syncthreads();

  // ---- phase 1: scores[m,k] = sum_d rst[m,d]*pf[m,k,d]  (4 col-tiles) ----
  for (int t = 0; t < 4; ++t) {
    CV acc;
#pragma unroll
    for (int r = 0; r < 8; ++r) acc.f[r] = 0.0f;
#pragma unroll
    for (int c = 0; c < 2; ++c) {
      AV a, b;
      // A: rst rows, 16x32 f16 layout (lane<16: K 0-15; lane>=16: K 16-31)
#pragma unroll
      for (int e = 0; e < 16; ++e) {
        int kl = e + hi * 16;
        a.h[e] = (_Float16)s_rst[wave][m16][32 * c + kl];
      }
      // B: col j=m16 -> (node 4t+(j>>2), k=j&3); K striped down elements
      int node = g + 4 * t + (m16 >> 2);
      if (node >= n_dst) node = n_dst - 1;
      const float* prow =
          pf + ((long long)node * HK + (m16 & 3)) * HD + 32 * c + hi * 16;
#pragma unroll
      for (int e = 0; e < 16; ++e) b.h[e] = (_Float16)prow[e];
      acc.v = __builtin_amdgcn_wmma_f32_16x16x32_f16(
          false, a.v, false, b.v, (short)0, acc.v, false, false);
    }
    // extract diagonal-block entries: need row m == col-node 4t+(j>>2)
#pragma unroll
    for (int r = 0; r < 8; ++r) {
      int mm = r + hi * 8;
      if ((mm >> 2) == t && (mm & 3) == (m16 >> 2))
        s_sc[wave][mm][m16 & 3] = acc.f[r];
    }
  }
  __syncthreads();

  // ---- softmax over K=4 per node (lanes 0..15 each own one node) ----
  if (lane < 16) {
    float s0 = s_sc[wave][lane][0] * 0.125f;
    float s1 = s_sc[wave][lane][1] * 0.125f;
    float s2 = s_sc[wave][lane][2] * 0.125f;
    float s3 = s_sc[wave][lane][3] * 0.125f;
    float mx = fmaxf(fmaxf(s0, s1), fmaxf(s2, s3));
    float e0 = __expf(s0 - mx), e1 = __expf(s1 - mx);
    float e2 = __expf(s2 - mx), e3 = __expf(s3 - mx);
    float inv = 1.0f / (e0 + e1 + e2 + e3);
    s_at[wave][lane][0] = (_Float16)(e0 * inv);
    s_at[wave][lane][1] = (_Float16)(e1 * inv);
    s_at[wave][lane][2] = (_Float16)(e2 * inv);
    s_at[wave][lane][3] = (_Float16)(e3 * inv);
  }
  __syncthreads();

  // ---- phase 2: pf_att = blockdiag(attn) x stacked pf, then finalize ----
  for (int tt = 0; tt < 4; ++tt) {
    CV acc;
#pragma unroll
    for (int r = 0; r < 8; ++r) acc.f[r] = 0.0f;
#pragma unroll
    for (int c = 0; c < 2; ++c) {
      AV a, b;
      // A row m16: nonzero only at j = 4*m16+k (block diagonal)
#pragma unroll
      for (int e = 0; e < 16; ++e) {
        int jc = e + hi * 16;
        int j  = 32 * c + jc;
        a.h[e] = ((j >> 2) == m16) ? s_at[wave][m16][j & 3] : (_Float16)0.0f;
      }
      // B row j = 4n+k holds pf[g+n, k, 16tt + col]
      int d0 = 16 * tt + m16;
#pragma unroll
      for (int e = 0; e < 16; ++e) {
        int j = 32 * c + hi * 16 + e;
        int node = g + (j >> 2);
        if (node >= n_dst) node = n_dst - 1;
        b.h[e] = (_Float16)pf[((long long)node * HK + (j & 3)) * HD + d0];
      }
      acc.v = __builtin_amdgcn_wmma_f32_16x16x32_f16(
          false, a.v, false, b.v, (short)0, acc.v, false, false);
    }
    // residual + tanh + degree norm; masked stores for tail
#pragma unroll
    for (int r = 0; r < 8; ++r) {
      int mm = r + hi * 8;
      int node = g + mm;
      int dd = 16 * tt + m16;
      if (node < n_dst) {
        float val =
            (s_rst[wave][mm][dd] + 0.1f * tanhf(acc.f[r])) * s_inv[wave][mm];
        rst[(long long)node * HD + dd] = val;
      }
    }
  }
}

// ---------------------------------------------------------------- launcher
extern "C" void kernel_launch(void* const* d_in, const int* in_sizes, int n_in,
                              void* d_out, int out_size, void* d_ws,
                              size_t ws_size, hipStream_t stream) {
  const float* feat_src = (const float*)d_in[0];
  // d_in[1] = feat_dst: unused by the reference math (only its shape matters)
  const float* pf = (const float*)d_in[2];
  const int* es = (const int*)d_in[3];
  const int* ed = (const int*)d_in[4];

  long long n_src = (long long)in_sizes[0] / HD;
  long long n_dst = (long long)in_sizes[1] / HD;
  long long nE    = (long long)in_sizes[3];

  float* out  = (float*)d_out;
  float* bsrc = out;                  // [n_src, 64]
  float* rst  = out + n_src * HD;     // [n_dst, 64]

  float* deg_dst = (float*)d_ws;      // [n_dst]
  float* deg_src = deg_dst + n_dst;   // [n_src]

  long long n_out_f = (n_src + n_dst) * HD;
  long long n_deg_f = n_src + n_dst;

  k_zero<<<2048, 256, 0, stream>>>(out, n_out_f, deg_dst, n_deg_f);

  unsigned eb = (unsigned)((nE * 16 + 255) / 256);
  k_fwd<<<eb, 256, 0, stream>>>(feat_src, es, ed, rst, deg_dst, deg_src, nE);

  unsigned ab = (unsigned)((n_dst + 127) / 128);
  k_attn<<<ab, 256, 0, stream>>>(rst, pf, deg_dst, (int)n_dst);

  k_bwd<<<eb, 256, 0, stream>>>(rst, es, ed, bsrc, nE);

  unsigned nb = (unsigned)((n_src * HD + 255) / 256);
  k_norm<<<nb, 256, 0, stream>>>(bsrc, deg_src, n_src * HD);
}
